// QwenAttention_heavy_hitter_44916767981649
// MI455X (gfx1250) — matrix operand order repaired and checked by reading, not compiled, not run on hardware
//
#include <hip/hip_runtime.h>
#include <hip/hip_bf16.h>

// ---------------- problem constants ----------------
#define S_LEN   2048
#define HID_DIM 3584
#define NH      28
#define KVH     4
#define HD      128
#define DQ      (NH * HD)    // 3584
#define DKV     (KVH * HD)   // 512
#define SCALE_F 0.08838834764831845f   // 128^-0.5
// H2O constants for kv_len = 2048
#define IPS     409          // image-protect prefix  (int(0.2*2048))
#define MSEL    409          // middle-select start
#define MESEL   1844         // middle-select end (2048 - 204 recent)
#define REGION  (MESEL - MSEL)   // 1435
#define KTOP    204          // heavy budget
#define NEG_INF (-3.0e38f)

typedef __bf16 v16bf __attribute__((ext_vector_type(16)));
typedef __bf16 v8bf  __attribute__((ext_vector_type(8)));
typedef float  v8f   __attribute__((ext_vector_type(8)));
typedef int    v4i   __attribute__((ext_vector_type(4)));

#define GLOBAL_AS __attribute__((address_space(1)))
#define LDS_AS    __attribute__((address_space(3)))

#if defined(__has_builtin)
#if __has_builtin(__builtin_amdgcn_global_load_async_to_lds_b128) && \
    __has_builtin(__builtin_amdgcn_s_wait_asynccnt)
#define HAVE_ASYNC_LDS 1
#endif
#endif
#ifndef HAVE_ASYNC_LDS
#define HAVE_ASYNC_LDS 0
#endif

// ---------------- helpers ----------------
static __device__ __forceinline__ unsigned short f2bf(float f) {
  unsigned u = __float_as_uint(f);
  u += 0x7FFFu + ((u >> 16) & 1u);      // round-to-nearest-even
  return (unsigned short)(u >> 16);
}

static __device__ __forceinline__ float wave_xor_f(float v, int m) {
  int lane = (int)(threadIdx.x & 31u);
  int idx  = ((lane ^ m) & 31) << 2;    // byte index for ds_bpermute (wave32)
  return __int_as_float(__builtin_amdgcn_ds_bpermute(idx, __float_as_int(v)));
}
static __device__ __forceinline__ float red16_max(float v) {
  v = fmaxf(v, wave_xor_f(v, 1)); v = fmaxf(v, wave_xor_f(v, 2));
  v = fmaxf(v, wave_xor_f(v, 4)); v = fmaxf(v, wave_xor_f(v, 8));
  return v;
}
static __device__ __forceinline__ float red16_sum(float v) {
  v += wave_xor_f(v, 1); v += wave_xor_f(v, 2);
  v += wave_xor_f(v, 4); v += wave_xor_f(v, 8);
  return v;
}

// A fragment: 16x32 bf16, row-major source (row = M, contiguous K).
// lane L: row = L&15; half = L>>4; K = half*8+{0..7} then 16+half*8+{0..7}
static __device__ __forceinline__ v16bf load_a32(const unsigned short* base, int ld,
                                                 int row0, int k0) {
  int lane = threadIdx.x & 31;
  int r = lane & 15, h = lane >> 4;
  const unsigned short* p = base + (size_t)(row0 + r) * ld + k0 + h * 8;
  v8bf lo = *(const v8bf*)p;
  v8bf hi = *(const v8bf*)(p + 16);
  return __builtin_shufflevector(lo, hi, 0,1,2,3,4,5,6,7,8,9,10,11,12,13,14,15);
}
// Same A fragment but from a 16x32 LDS tile (row stride 32 elements).
static __device__ __forceinline__ v16bf lds_a_frag(const unsigned short* s) {
  int lane = threadIdx.x & 31;
  int r = lane & 15, h = lane >> 4;
  const unsigned short* p = s + r * 32 + h * 8;
  v8bf lo = *(const v8bf*)p;
  v8bf hi = *(const v8bf*)(p + 16);
  return __builtin_shufflevector(lo, hi, 0,1,2,3,4,5,6,7,8,9,10,11,12,13,14,15);
}
// B fragment: 32x16 bf16 where B[k][n] = base[(col0+n)*ld + k0 + k].
// lane L: col = L&15; half = L>>4; K = half*8+{0..7} then 16+half*8+{0..7}
static __device__ __forceinline__ v16bf load_b32(const unsigned short* base, int ld,
                                                 int col0, int k0) {
  int lane = threadIdx.x & 31;
  int c = lane & 15, h = lane >> 4;
  const unsigned short* p = base + (size_t)(col0 + c) * ld + k0 + h * 8;
  v8bf lo = *(const v8bf*)p;
  v8bf hi = *(const v8bf*)(p + 16);
  return __builtin_shufflevector(lo, hi, 0,1,2,3,4,5,6,7,8,9,10,11,12,13,14,15);
}

// 16B global -> LDS copy: CDNA5 async path (ASYNCcnt) with portable fallback.
static __device__ __forceinline__ void copy16_issue(const unsigned short* g, unsigned short* l) {
#if HAVE_ASYNC_LDS
  unsigned short* gnc = const_cast<unsigned short*>(g);
  GLOBAL_AS v4i* gp = (GLOBAL_AS v4i*)gnc;
  LDS_AS    v4i* lp = (LDS_AS v4i*)l;
  __builtin_amdgcn_global_load_async_to_lds_b128(gp, lp, 0, 0);
#else
  *(v8bf*)l = *(const v8bf*)g;
#endif
}
static __device__ __forceinline__ void copy16_wait() {
#if HAVE_ASYNC_LDS
  __builtin_amdgcn_s_wait_asynccnt(0);
#endif
}

// ---------------- fp32 -> bf16 conversion ----------------
__global__ void cvt_bf16_kernel(const float* __restrict__ in,
                                unsigned short* __restrict__ outp, int n) {
  int i = blockIdx.x * blockDim.x + threadIdx.x;
  if (i < n) outp[i] = f2bf(in[i]);
}

// ---------------- WMMA GEMM: C[m,n] = sum_k A[m,k]*W[n,k] ----------------
// A bf16 row-major [M,K] (lda), W bf16 row-major [N,K] (ldw), C fp32 (ldc).
// Block tile 32x512 (8 waves, each 32x64). A tile staged in LDS via async
// global->LDS double buffer; B fragments double-buffered in registers so the
// s_wait on loads lands one iteration after issue.
__global__ __launch_bounds__(256) void gemm_bf16_kernel(
    const unsigned short* __restrict__ A, int lda,
    const unsigned short* __restrict__ W, int ldw,
    float* __restrict__ C, int ldc, int K) {
  __shared__ __attribute__((aligned(16))) unsigned short Astage[2][32][32];
  int wid  = threadIdx.x >> 5;
  int lane = threadIdx.x & 31;
  int tid  = threadIdx.x;
  int m0 = blockIdx.x * 32;
  int n0 = blockIdx.y * 512 + wid * 64;

  // A-tile copier: 32x32 bf16 = 128 x 16B chunks; waves 0-3 (tid<128) own one each.
  bool copier = (tid < 128);
  int crow = tid >> 2;
  int ck   = (tid & 3) * 8;
  const unsigned short* Acp = A + (size_t)(m0 + crow) * lda + ck;

  if (copier) copy16_issue(Acp, &Astage[0][crow][ck]);
  if (wid < 4) copy16_wait();
  __syncthreads();

  v8f acc[8] = {};              // [mi*4 + nt]
  v16bf bc[4];
#pragma unroll
  for (int t = 0; t < 4; t++) bc[t] = load_b32(W, ldw, n0 + t * 16, 0);

  for (int k0 = 0; k0 < K; k0 += 32) {
    int buf = (k0 >> 5) & 1;
    bool last = (k0 + 32 >= K);
    if (!last && copier)
      copy16_issue(Acp + k0 + 32, &Astage[buf ^ 1][crow][ck]);

    v16bf a0 = lds_a_frag(&Astage[buf][0][0]);
    v16bf a1 = lds_a_frag(&Astage[buf][16][0]);

    v16bf bn[4];
    if (!last) {
#pragma unroll
      for (int t = 0; t < 4; t++) bn[t] = load_b32(W, ldw, n0 + t * 16, k0 + 32);
    }
#pragma unroll
    for (int t = 0; t < 4; t++) {
      acc[t]     = __builtin_amdgcn_wmma_f32_16x16x32_bf16(false, a0, false, bc[t], (short)0, acc[t],     false, false);
      acc[4 + t] = __builtin_amdgcn_wmma_f32_16x16x32_bf16(false, a1, false, bc[t], (short)0, acc[4 + t], false, false);
    }
    if (!last) {
      if (wid < 4) copy16_wait();     // async A for next buf complete
      __syncthreads();                // publish LDS + protect buffer reuse
#pragma unroll
      for (int t = 0; t < 4; t++) bc[t] = bn[t];
    }
  }
  int c = lane & 15, h = lane >> 4;
#pragma unroll
  for (int mi = 0; mi < 2; mi++)
#pragma unroll
    for (int t = 0; t < 4; t++)
#pragma unroll
      for (int r = 0; r < 8; r++)
        C[(size_t)(m0 + mi * 16 + r + 8 * h) * ldc + n0 + t * 16 + c] = acc[mi * 4 + t][r];
}

// ---------------- RoPE + bias + bf16 pack (mrope collapses to plain rope) ----------------
__global__ void rope_kernel(const float* __restrict__ t, const float* __restrict__ bias,
                            const float* __restrict__ cosb, const float* __restrict__ sinb,
                            unsigned short* __restrict__ outp, int nheads, int rowdim) {
  int idx = blockIdx.x * blockDim.x + threadIdx.x;
  int total = nheads * S_LEN * 64;
  if (idx >= total) return;
  int d  = idx & 63;
  int s  = (idx >> 6) & (S_LEN - 1);
  int hh = idx >> 17;                       // /(S_LEN*64)
  int c0i = hh * HD + d, c1i = c0i + 64;
  float x0 = t[(size_t)s * rowdim + c0i] + bias[c0i];
  float x1 = t[(size_t)s * rowdim + c1i] + bias[c1i];
  float c0 = cosb[(size_t)s * HD + d],      s0 = sinb[(size_t)s * HD + d];
  float c1 = cosb[(size_t)s * HD + d + 64], s1 = sinb[(size_t)s * HD + d + 64];
  float e0 = x0 * c0 - x1 * s0;             // rotate_half: [-x_hi, x_lo]
  float e1 = x1 * c1 + x0 * s1;
  size_t ob = ((size_t)hh * S_LEN + s) * HD;
  outp[ob + d]      = f2bf(e0);
  outp[ob + d + 64] = f2bf(e1);
}

// V: bias + bf16, transposed to [kv][d][s] so P@V B-fragments are K-contiguous
__global__ void convv_kernel(const float* __restrict__ vtmp, const float* __restrict__ vb,
                             unsigned short* __restrict__ Vt) {
  int idx = blockIdx.x * blockDim.x + threadIdx.x;
  int total = KVH * S_LEN * HD;
  if (idx >= total) return;
  int d  = idx & 127;
  int s  = (idx >> 7) & (S_LEN - 1);
  int kv = idx >> 18;
  float val = vtmp[(size_t)s * DKV + kv * HD + d] + vb[kv * HD + d];
  Vt[((size_t)kv * HD + d) * S_LEN + s] = f2bf(val);
}

// ---------------- attention pass 1: softmax stats + H2O key scores ----------------
// grid (32, 28), block 128 (4 waves); wave = one 16-row q tile of one head.
// K-tile B fragments are register double-buffered to hide load latency.
__global__ __launch_bounds__(128) void attn_pass1_kernel(
    const unsigned short* __restrict__ Qa, const unsigned short* __restrict__ Ka,
    float* __restrict__ m_g, float* __restrict__ l_g, float* __restrict__ scores) {
  int wid = threadIdx.x >> 5, lane = threadIdx.x & 31;
  int h = blockIdx.y, kv = h / (NH / KVH);
  int qtile = blockIdx.x * 4 + wid, q0 = qtile * 16;
  const unsigned short* Qh = Qa + (size_t)h  * S_LEN * HD;
  const unsigned short* Kh = Ka + (size_t)kv * S_LEN * HD;
  int n = lane & 15, half = lane >> 4;

  v16bf aq[4];
#pragma unroll
  for (int kk = 0; kk < 4; kk++) aq[kk] = load_a32(Qh, HD, q0, kk * 32);

  float m8[8], l8[8];
#pragma unroll
  for (int r = 0; r < 8; r++) { m8[r] = NEG_INF; l8[r] = 0.0f; }

  // ---- sub-pass A: online row max / row sum over causal range ----
  {
    v16bf bt[4];
#pragma unroll
    for (int kk = 0; kk < 4; kk++) bt[kk] = load_b32(Kh, HD, 0, kk * 32);
    for (int kt = 0; kt <= qtile; kt++) {
      v16bf bn[4];
      if (kt < qtile) {
#pragma unroll
        for (int kk = 0; kk < 4; kk++) bn[kk] = load_b32(Kh, HD, (kt + 1) * 16, kk * 32);
      }
      v8f sc = {};
#pragma unroll
      for (int kk = 0; kk < 4; kk++)
        sc = __builtin_amdgcn_wmma_f32_16x16x32_bf16(false, aq[kk], false, bt[kk], (short)0, sc, false, false);
      int j = kt * 16 + n;
      bool diag = (kt == qtile);
#pragma unroll
      for (int r = 0; r < 8; r++) {
        int i = q0 + r + 8 * half;
        bool ok = (!diag) || (j <= i);
        float s = ok ? sc[r] * SCALE_F : NEG_INF;
        float mn = fmaxf(m8[r], red16_max(s));
        float e  = ok ? __expf(s - mn) : 0.0f;
        float es = red16_sum(e);
        l8[r] = l8[r] * __expf(m8[r] - mn) + es;
        m8[r] = mn;
      }
      if (kt < qtile) {
#pragma unroll
        for (int kk = 0; kk < 4; kk++) bt[kk] = bn[kk];
      }
    }
  }
  if (n == 0) {
#pragma unroll
    for (int r = 0; r < 8; r++) {
      m_g[(size_t)h * S_LEN + q0 + r + 8 * half] = m8[r];
      l_g[(size_t)h * S_LEN + q0 + r + 8 * half] = l8[r];
    }
  }
  float rl8[8];
#pragma unroll
  for (int r = 0; r < 8; r++) rl8[r] = 1.0f / l8[r];

  // ---- sub-pass B: accumulate per-key post-softmax mass (H2O scores) ----
  {
    v16bf bt[4];
#pragma unroll
    for (int kk = 0; kk < 4; kk++) bt[kk] = load_b32(Kh, HD, 0, kk * 32);
    for (int kt = 0; kt <= qtile; kt++) {
      v16bf bn[4];
      if (kt < qtile) {
#pragma unroll
        for (int kk = 0; kk < 4; kk++) bn[kk] = load_b32(Kh, HD, (kt + 1) * 16, kk * 32);
      }
      v8f sc = {};
#pragma unroll
      for (int kk = 0; kk < 4; kk++)
        sc = __builtin_amdgcn_wmma_f32_16x16x32_bf16(false, aq[kk], false, bt[kk], (short)0, sc, false, false);
      int j = kt * 16 + n;
      bool diag = (kt == qtile);
      float csum = 0.0f;
#pragma unroll
      for (int r = 0; r < 8; r++) {
        int i = q0 + r + 8 * half;
        bool ok = (!diag) || (j <= i);
        float p = ok ? __expf(sc[r] * SCALE_F - m8[r]) * rl8[r] : 0.0f;
        csum += p;
      }
      csum += wave_xor_f(csum, 16);   // rows 0-7 + rows 8-15 (same column)
      if (lane < 16) atomicAdd(scores + (size_t)h * S_LEN + kt * 16 + lane, csum);
      if (kt < qtile) {
#pragma unroll
        for (int kk = 0; kk < 4; kk++) bt[kk] = bn[kk];
      }
    }
  }
}

// ---------------- H2O top-k keep-mask (binary select on float bits) ----------------
__global__ __launch_bounds__(256) void h2o_topk_kernel(const float* __restrict__ scores,
                                                       unsigned char* __restrict__ keep) {
  __shared__ unsigned int keys[REGION];
  __shared__ int red[256];
  __shared__ int total_gt, ctr;
  int h = blockIdx.x, tid = threadIdx.x;
  const float* sh = scores + (size_t)h * S_LEN;
  unsigned char* kh = keep + (size_t)h * S_LEN;
  for (int i = tid; i < REGION; i += 256) keys[i] = __float_as_uint(sh[MSEL + i]); // scores >= 0
  for (int i = tid; i < IPS; i += 256) kh[i] = 1;                  // sink / image prefix
  for (int i = MESEL + tid; i < S_LEN; i += 256) kh[i] = 1;        // recent window
  __syncthreads();

  unsigned int lo = 0u, hi = 0xFFFFFFFFu;   // find k-th largest key value
  while (lo < hi) {
    unsigned int mid = lo + ((hi - lo) >> 1) + 1u;
    int c = 0;
    for (int i = tid; i < REGION; i += 256) c += (keys[i] >= mid) ? 1 : 0;
    red[tid] = c; __syncthreads();
    for (int s2 = 128; s2 > 0; s2 >>= 1) { if (tid < s2) red[tid] += red[tid + s2]; __syncthreads(); }
    int cnt = red[0]; __syncthreads();
    if (cnt >= KTOP) lo = mid; else hi = mid - 1u;
  }
  unsigned int thr = lo;
  int c = 0;
  for (int i = tid; i < REGION; i += 256) c += (keys[i] > thr) ? 1 : 0;
  red[tid] = c; __syncthreads();
  for (int s2 = 128; s2 > 0; s2 >>= 1) { if (tid < s2) red[tid] += red[tid + s2]; __syncthreads(); }
  if (tid == 0) { total_gt = red[0]; ctr = 0; }
  __syncthreads();
  int quota = KTOP - total_gt;
  for (int i = tid; i < REGION; i += 256) {
    unsigned int kvv = keys[i];
    if (kvv > thr) kh[MSEL + i] = 1;
    else if (kvv == thr) {
      int p = atomicAdd(&ctr, 1);
      if (p < quota) kh[MSEL + i] = 1;
    }
  }
}

// ---------------- attention pass 2: masked renormalized P @ V ----------------
__global__ __launch_bounds__(128) void attn_pass2_kernel(
    const unsigned short* __restrict__ Qa, const unsigned short* __restrict__ Ka,
    const unsigned short* __restrict__ Vt, const float* __restrict__ m_g,
    const float* __restrict__ l_g, const unsigned char* __restrict__ keep,
    unsigned short* __restrict__ attn) {
  __shared__ __attribute__((aligned(32))) unsigned short stage[4][16][32];
  int wid = threadIdx.x >> 5, lane = threadIdx.x & 31;
  int h = blockIdx.y, kv = h / (NH / KVH);
  int qtile = blockIdx.x * 4 + wid, q0 = qtile * 16;
  const unsigned short* Qh = Qa + (size_t)h  * S_LEN * HD;
  const unsigned short* Kh = Ka + (size_t)kv * S_LEN * HD;
  const unsigned short* Vh = Vt + (size_t)kv * HD * S_LEN;
  int n = lane & 15, half = lane >> 4;

  v16bf aq[4];
#pragma unroll
  for (int kk = 0; kk < 4; kk++) aq[kk] = load_a32(Qh, HD, q0, kk * 32);

  float m8[8], l8[8], lp8[8];
#pragma unroll
  for (int r = 0; r < 8; r++) {
    m8[r]  = m_g[(size_t)h * S_LEN + q0 + r + 8 * half];
    l8[r]  = l_g[(size_t)h * S_LEN + q0 + r + 8 * half];
    lp8[r] = 0.0f;
  }
  v8f acc[8] = {};

  for (int kt0 = 0; kt0 <= qtile; kt0 += 2) {
#pragma unroll
    for (int t = 0; t < 2; t++) {
      int kt = kt0 + t;
      float pv[8];
      if (kt <= qtile) {
        v8f sc = {};
#pragma unroll
        for (int kk = 0; kk < 4; kk++) {
          v16bf b = load_b32(Kh, HD, kt * 16, kk * 32);
          sc = __builtin_amdgcn_wmma_f32_16x16x32_bf16(false, aq[kk], false, b, (short)0, sc, false, false);
        }
        int j = kt * 16 + n;
        float kp = (float)keep[(size_t)h * S_LEN + j];
        bool diag = (kt == qtile);
#pragma unroll
        for (int r = 0; r < 8; r++) {
          int i = q0 + r + 8 * half;
          bool ok = (!diag) || (j <= i);
          float p = ok ? __expf(sc[r] * SCALE_F - m8[r]) * kp : 0.0f;
          pv[r] = p;
          lp8[r] += red16_sum(p);
        }
      } else {
#pragma unroll
        for (int r = 0; r < 8; r++) pv[r] = 0.0f;
      }
#pragma unroll
      for (int r = 0; r < 8; r++)                     // C-layout -> LDS (16x32 tile)
        stage[wid][r + 8 * half][t * 16 + n] = f2bf(pv[r]);
    }
    // re-read as A-fragment (per-wave LDS region; compiler orders via DScnt)
    const unsigned short* sp = &stage[wid][lane & 15][half * 8];
    v8bf lo = *(const v8bf*)sp;
    v8bf hi = *(const v8bf*)(sp + 16);
    v16bf pa = __builtin_shufflevector(lo, hi, 0,1,2,3,4,5,6,7,8,9,10,11,12,13,14,15);
#pragma unroll
    for (int dt = 0; dt < 8; dt++) {
      v16bf bv = load_b32(Vh, S_LEN, dt * 16, kt0 * 16);   // V^T: K(keys) contiguous
      acc[dt] = __builtin_amdgcn_wmma_f32_16x16x32_bf16(false, pa, false, bv, (short)0, acc[dt], false, false);
    }
  }
  float dn[8];
#pragma unroll
  for (int r = 0; r < 8; r++) dn[r] = 1.0f / fmaxf(lp8[r], 1e-9f * l8[r]);
#pragma unroll
  for (int dt = 0; dt < 8; dt++)
#pragma unroll
    for (int r = 0; r < 8; r++)
      attn[(size_t)(q0 + r + 8 * half) * DQ + h * HD + dt * 16 + n] = f2bf(acc[dt][r] * dn[r]);
}

// ---------------- host-side orchestration ----------------
extern "C" void kernel_launch(void* const* d_in, const int* in_sizes, int n_in,
                              void* d_out, int out_size, void* d_ws, size_t ws_size,
                              hipStream_t stream) {
  (void)in_sizes; (void)n_in; (void)out_size; (void)ws_size;
  const float* hs   = (const float*)d_in[0];
  /* d_in[1] = attention_mask: causal, modeled analytically */
  const float* cosb = (const float*)d_in[2];   // [3,B,S,128] -> axis0=0 slice
  const float* sinb = (const float*)d_in[3];
  const float* q_w  = (const float*)d_in[4];
  const float* q_b  = (const float*)d_in[5];
  const float* k_w  = (const float*)d_in[6];
  const float* k_b  = (const float*)d_in[7];
  const float* v_w  = (const float*)d_in[8];
  const float* v_b  = (const float*)d_in[9];
  const float* o_w  = (const float*)d_in[10];
  float* outp = (float*)d_out;

  char* ws = (char*)d_ws;
  size_t off = 0;
  auto alloc = [&](size_t bytes) -> char* {
    char* p = ws + off;
    off = (off + bytes + 255) & ~(size_t)255;
    return p;
  };
  unsigned short* hsb   = (unsigned short*)alloc((size_t)S_LEN * HID_DIM * 2);
  unsigned short* qwb   = (unsigned short*)alloc((size_t)DQ * HID_DIM * 2);
  unsigned short* kwb   = (unsigned short*)alloc((size_t)DKV * HID_DIM * 2);
  unsigned short* vwb   = (unsigned short*)alloc((size_t)DKV * HID_DIM * 2);
  unsigned short* owb   = (unsigned short*)alloc((size_t)HID_DIM * DQ * 2);
  float* qtmp           = (float*)alloc((size_t)S_LEN * DQ * 4);
  float* ktmp           = (float*)alloc((size_t)S_LEN * DKV * 4);
  float* vtmp           = (float*)alloc((size_t)S_LEN * DKV * 4);
  unsigned short* Qa    = (unsigned short*)alloc((size_t)NH  * S_LEN * HD * 2);
  unsigned short* Ka    = (unsigned short*)alloc((size_t)KVH * S_LEN * HD * 2);
  unsigned short* Vt    = (unsigned short*)alloc((size_t)KVH * HD * S_LEN * 2);
  float* m_g            = (float*)alloc((size_t)NH * S_LEN * 4);
  float* l_g            = (float*)alloc((size_t)NH * S_LEN * 4);
  float* scores         = (float*)alloc((size_t)NH * S_LEN * 4);
  unsigned char* keepm  = (unsigned char*)alloc((size_t)NH * S_LEN);
  unsigned short* attn  = (unsigned short*)qtmp;   // reuse: qtmp dead after RoPE

  (void)hipMemsetAsync(scores, 0, (size_t)NH * S_LEN * 4, stream);
  (void)hipMemsetAsync(keepm, 0, (size_t)NH * S_LEN, stream);

  auto cvt = [&](const float* src, unsigned short* dst, int nelem) {
    cvt_bf16_kernel<<<(nelem + 255) / 256, 256, 0, stream>>>(src, dst, nelem);
  };
  cvt(hs,  hsb, S_LEN * HID_DIM);
  cvt(q_w, qwb, DQ * HID_DIM);
  cvt(k_w, kwb, DKV * HID_DIM);
  cvt(v_w, vwb, DKV * HID_DIM);
  cvt(o_w, owb, HID_DIM * DQ);

  // QKV projections (bias folded into RoPE / V-convert epilogues)
  gemm_bf16_kernel<<<dim3(S_LEN / 32, DQ  / 512), 256, 0, stream>>>(hsb, HID_DIM, qwb, HID_DIM, qtmp, DQ,  HID_DIM);
  gemm_bf16_kernel<<<dim3(S_LEN / 32, DKV / 512), 256, 0, stream>>>(hsb, HID_DIM, kwb, HID_DIM, ktmp, DKV, HID_DIM);
  gemm_bf16_kernel<<<dim3(S_LEN / 32, DKV / 512), 256, 0, stream>>>(hsb, HID_DIM, vwb, HID_DIM, vtmp, DKV, HID_DIM);

  rope_kernel<<<(NH  * S_LEN * 64 + 255) / 256, 256, 0, stream>>>(qtmp, q_b, cosb, sinb, Qa, NH,  DQ);
  rope_kernel<<<(KVH * S_LEN * 64 + 255) / 256, 256, 0, stream>>>(ktmp, k_b, cosb, sinb, Ka, KVH, DKV);
  convv_kernel<<<(KVH * S_LEN * HD + 255) / 256, 256, 0, stream>>>(vtmp, v_b, Vt);

  attn_pass1_kernel<<<dim3(S_LEN / 64, NH), 128, 0, stream>>>(Qa, Ka, m_g, l_g, scores);
  h2o_topk_kernel<<<NH, 256, 0, stream>>>(scores, keepm);
  attn_pass2_kernel<<<dim3(S_LEN / 64, NH), 128, 0, stream>>>(Qa, Ka, Vt, m_g, l_g, keepm, attn);

  // output projection -> fp32 result
  gemm_bf16_kernel<<<dim3(S_LEN / 32, HID_DIM / 512), 256, 0, stream>>>(attn, DQ, owb, DQ, outp, HID_DIM, DQ);
}
